// cross_attention_34849364639877
// MI455X (gfx1250) — matrix operand order repaired and checked by reading, not compiled, not run on hardware
//
#include <hip/hip_runtime.h>
#include <hip/hip_bf16.h>
#include <cstdint>

typedef __attribute__((ext_vector_type(16))) _Float16 v16h;
typedef __attribute__((ext_vector_type(8)))  _Float16 v8h;
typedef __attribute__((ext_vector_type(4)))  _Float16 v4h;
typedef __attribute__((ext_vector_type(8)))  float    v8f;
typedef __attribute__((ext_vector_type(4)))  int      v4i;

#define NUM_HEADS 16
#define EMBED_DIM 1024
#define CROSS_DIM 768
#define HEAD_DIM  64
#define BATCH     4
#define SEQ       1024
#define BS        (BATCH * SEQ)

// ---------------------------------------------------------------------------
// CDNA5 async global->LDS copy (ASYNCcnt path), with synchronous fallback.
// Builtin signature (from hipcc diagnostic): param 1 is v4i in AS(1)
// (printed as "int __vector(4) __device__ *"); LDS side is AS(3).
// ---------------------------------------------------------------------------
typedef __attribute__((address_space(1))) v4i as1_v4i;
typedef __attribute__((address_space(3))) v4i as3_v4i;

#if defined(__has_builtin)
#if __has_builtin(__builtin_amdgcn_global_load_async_to_lds_b128)
#define HAVE_ASYNC_LDS 1
#endif
#endif
#ifndef HAVE_ASYNC_LDS
#define HAVE_ASYNC_LDS 0
#endif

__device__ __forceinline__ void async_b128(const _Float16* g, _Float16* l) {
#if HAVE_ASYNC_LDS
  __builtin_amdgcn_global_load_async_to_lds_b128((as1_v4i*)g, (as3_v4i*)l, 0,
                                                 0);
#else
  *(v8h*)l = *(const v8h*)g;
#endif
}
__device__ __forceinline__ void async_b256(const _Float16* g, _Float16* l) {
  async_b128(g, l);
  async_b128(g + 8, l + 8);
}
__device__ __forceinline__ void wait_async() {
#if HAVE_ASYNC_LDS
#if __has_builtin(__builtin_amdgcn_s_wait_asynccnt)
  __builtin_amdgcn_s_wait_asynccnt(0);
#else
  asm volatile("s_wait_asynccnt 0" ::: "memory");
#endif
#endif
}

__device__ __forceinline__ v16h cat8(v8h lo, v8h hi) {
  return __builtin_shufflevector(lo, hi, 0,1,2,3,4,5,6,7,8,9,10,11,12,13,14,15);
}

__device__ __forceinline__ v8f wmma16(v16h a, v16h b, v8f c) {
  // D = A(16x32 f16) x B(32x16 f16) + C(16x16 f32)
  return __builtin_amdgcn_wmma_f32_16x16x32_f16(false, a, false, b, (short)0, c,
                                                false, false);
}

// ---------------------------------------------------------------------------
// f32 -> f16 conversion (vectorized, grid-stride)
// ---------------------------------------------------------------------------
__global__ __launch_bounds__(256) void cvt_f32_f16_kernel(
    const float4* __restrict__ in, _Float16* __restrict__ out, int n4) {
  int i = blockIdx.x * 256 + threadIdx.x;
  int stride = gridDim.x * 256;
  for (; i < n4; i += stride) {
    float4 v = in[i];
    v4h h = { (_Float16)v.x, (_Float16)v.y, (_Float16)v.z, (_Float16)v.w };
    *(v4h*)(out + (size_t)i * 4) = h;
  }
}

// ---------------------------------------------------------------------------
// NT GEMM: C[M,N] = sum_k A[M,K]*B[N,K] + bias[N]
// A row-major [M,K] f16, B row-major [N,K] f16 (torch Linear weight layout).
// Block: 256 threads (8 waves), tile 128(M) x 128(N), K chunked by 32.
// Wave (wm 0..3, wn 0..1) computes a 32x64 tile = 2x4 WMMA tiles.
// Double-buffered LDS; chunk k+1 streams in via async-LDS during chunk-k WMMAs.
// ---------------------------------------------------------------------------
template <bool OUT16>
__global__ __launch_bounds__(256) void gemm_nt_wmma_kernel(
    const _Float16* __restrict__ A, const _Float16* __restrict__ Bw,
    const float* __restrict__ bias, _Float16* __restrict__ out16,
    float* __restrict__ out32, int M, int N, int K) {
  __shared__ alignas(32) _Float16 As[2][128][32];
  __shared__ alignas(32) _Float16 Bs[2][128][32];

  const int t = threadIdx.x;
  const int lane = t & 31;
  const int wave = t >> 5;
  const int wm = wave >> 1;      // 0..3 (M)
  const int wn = wave & 1;       // 0..1 (N)
  const int m0 = blockIdx.x * 128;
  const int n0 = blockIdx.y * 128;
  const int lrow  = lane & 15;
  const int lhalf = lane >> 4;

  // staging: each thread moves one 16-half (32B) chunk of A and of B
  const int srow  = t >> 1;           // 0..127
  const int skoff = (t & 1) * 16;     // 0 or 16

  const _Float16* ag = &A[(size_t)(m0 + srow) * K + skoff];
  const _Float16* bg = &Bw[(size_t)(n0 + srow) * K + skoff];

  async_b256(ag, &As[0][srow][skoff]);    // chunk 0 in flight
  async_b256(bg, &Bs[0][srow][skoff]);

  const v8f vzero = {};
  v8f acc[2][4];
#pragma unroll
  for (int i = 0; i < 2; ++i)
#pragma unroll
    for (int j = 0; j < 4; ++j) acc[i][j] = vzero;

  int buf = 0;
  for (int kc = 0; kc < K; kc += 32) {
    wait_async();
    __syncthreads();                       // chunk kc resident in buf
    if (kc + 32 < K) {                     // uniform: stream next chunk
      async_b256(ag + kc + 32, &As[buf ^ 1][srow][skoff]);
      async_b256(bg + kc + 32, &Bs[buf ^ 1][srow][skoff]);
    }

    v16h af[2], bf[4];
#pragma unroll
    for (int i = 0; i < 2; ++i) {
      const int r = wm * 32 + i * 16 + lrow;
      af[i] = cat8(*(const v8h*)&As[buf][r][lhalf * 8],
                   *(const v8h*)&As[buf][r][lhalf * 8 + 16]);
    }
#pragma unroll
    for (int j = 0; j < 4; ++j) {
      const int r = wn * 64 + j * 16 + lrow;
      bf[j] = cat8(*(const v8h*)&Bs[buf][r][lhalf * 8],
                   *(const v8h*)&Bs[buf][r][lhalf * 8 + 16]);
    }
#pragma unroll
    for (int i = 0; i < 2; ++i)
#pragma unroll
      for (int j = 0; j < 4; ++j) acc[i][j] = wmma16(af[i], bf[j], acc[i][j]);
    buf ^= 1;
  }

  // Epilogue: C layout -> element e of lane = (M = e + 8*lhalf, N = lrow)
#pragma unroll
  for (int j = 0; j < 4; ++j) {
    const int n = n0 + wn * 64 + j * 16 + lrow;
    const float bv = bias[n];
#pragma unroll
    for (int i = 0; i < 2; ++i) {
      const int mbase = m0 + wm * 32 + i * 16 + lhalf * 8;
#pragma unroll
      for (int e = 0; e < 8; ++e) {
        const float v = acc[i][j][e] + bv;
        const size_t off = (size_t)(mbase + e) * N + n;
        if (OUT16) out16[off] = (_Float16)v;
        else       out32[off] = v;
      }
    }
  }
}

// ---------------------------------------------------------------------------
// Fused flash attention. Workgroup: one (b,h), 128 queries (8 waves x 16 q).
// Double-buffered K (async-LDS) and V^T (register-prefetched transpose).
// Online softmax with 16-lane butterfly reductions in the WMMA C layout.
//   qh  : [B,S,EMBED] f16   kvh : [B,S,2*EMBED] f16   oh : [B,S,EMBED] f16
// ---------------------------------------------------------------------------
__global__ __launch_bounds__(256) void attn_fa_wmma_kernel(
    const _Float16* __restrict__ qh, const _Float16* __restrict__ kvh,
    _Float16* __restrict__ oh) {
  __shared__ alignas(32) _Float16 Ks[2][32][64];    // [buf][key][d]
  __shared__ alignas(32) _Float16 Vt[2][64][32];    // [buf][d][key]
  __shared__ alignas(32) _Float16 Ps[8][16][32];    // per-wave P scratch

  const int t = threadIdx.x;
  const int lane = t & 31;
  const int wave = t >> 5;
  const int lrow  = lane & 15;
  const int lhalf = lane >> 4;
  const int b = blockIdx.y >> 4;
  const int h = blockIdx.y & 15;
  const int q0 = blockIdx.x * 128 + wave * 16;

  const size_t qbase  = (size_t)b * SEQ * EMBED_DIM;
  const size_t kvbase = (size_t)b * SEQ * (2 * EMBED_DIM);

  // Q fragments for the two 32-wide d chunks, resident all loop.
  v16h qf[2];
#pragma unroll
  for (int c = 0; c < 2; ++c) {
    const _Float16* qp = qh + qbase + (size_t)(q0 + lrow) * EMBED_DIM +
                         h * HEAD_DIM + c * 32 + lhalf * 8;
    qf[c] = cat8(*(const v8h*)qp, *(const v8h*)(qp + 16));
  }

  const v8f vzero = {};
  v8f o[4];
#pragma unroll
  for (int nd = 0; nd < 4; ++nd) o[nd] = vzero;
  float mrow[8], lsum[8];
#pragma unroll
  for (int e = 0; e < 8; ++e) { mrow[e] = -1e30f; lsum[e] = 0.0f; }

  const int skey = t >> 3;        // 0..31
  const int sd   = (t & 7) * 8;   // 0,8,..,56
  const _Float16* kg = kvh + kvbase + h * HEAD_DIM + sd;  // K source
  const _Float16* vg = kg + EMBED_DIM;                    // V source

  // prologue: chunk 0 — K via async, V into registers
  async_b128(kg + (size_t)skey * (2 * EMBED_DIM), &Ks[0][skey][sd]);
  v8h vreg = *(const v8h*)(vg + (size_t)skey * (2 * EMBED_DIM));

  int buf = 0;
  for (int kc = 0; kc < SEQ; kc += 32) {
    // transpose-store current V chunk (buffer free since iter kc-2)
#pragma unroll
    for (int e = 0; e < 8; ++e) Vt[buf][sd + e][skey] = vreg[e];
    wait_async();
    __syncthreads();                 // Ks[buf], Vt[buf] ready for all waves
    if (kc + 32 < SEQ) {             // stream next chunk during compute
      async_b128(kg + (size_t)(kc + 32 + skey) * (2 * EMBED_DIM),
                 &Ks[buf ^ 1][skey][sd]);
      vreg = *(const v8h*)(vg + (size_t)(kc + 32 + skey) * (2 * EMBED_DIM));
    }

    // scores: two 16x16 tiles (keys kc..kc+15, kc+16..kc+31)
    v8f sc[2];
#pragma unroll
    for (int n = 0; n < 2; ++n) {
      v8f c = vzero;
#pragma unroll
      for (int dc = 0; dc < 2; ++dc) {
        v16h kf = cat8(*(const v8h*)&Ks[buf][n * 16 + lrow][dc * 32 + lhalf * 8],
                       *(const v8h*)&Ks[buf][n * 16 + lrow][dc * 32 + lhalf * 8 + 16]);
        c = wmma16(qf[dc], kf, c);
      }
      sc[n] = c;
    }

    // online softmax; row of element e lives across the lane's 16-lane half
    float alpha[8];
#pragma unroll
    for (int e = 0; e < 8; ++e) {
      float s0 = sc[0][e] * 0.125f;   // 1/sqrt(64)
      float s1 = sc[1][e] * 0.125f;
      float cm = fmaxf(s0, s1);
#pragma unroll
      for (int msk = 8; msk >= 1; msk >>= 1)
        cm = fmaxf(cm, __shfl_xor(cm, msk, 32));
      const float mn = fmaxf(mrow[e], cm);
      const float al = __expf(mrow[e] - mn);
      const float p0 = __expf(s0 - mn);
      const float p1 = __expf(s1 - mn);
      float rs = p0 + p1;
#pragma unroll
      for (int msk = 8; msk >= 1; msk >>= 1) rs += __shfl_xor(rs, msk, 32);
      lsum[e] = lsum[e] * al + rs;
      mrow[e] = mn;
      alpha[e] = al;
      Ps[wave][e + 8 * lhalf][lrow]      = (_Float16)p0;
      Ps[wave][e + 8 * lhalf][16 + lrow] = (_Float16)p1;
    }
#pragma unroll
    for (int nd = 0; nd < 4; ++nd)
#pragma unroll
      for (int e = 0; e < 8; ++e) o[nd][e] *= alpha[e];

    __syncthreads();  // order Ps stores before fragment reloads

    // P (16x32) as A fragment, accumulate P @ V into o
    v16h pf;
    {
      v8h lo = *(const v8h*)&Ps[wave][lrow][lhalf * 8];
      v8h hi = *(const v8h*)&Ps[wave][lrow][lhalf * 8 + 16];
      pf = cat8(lo, hi);
    }
#pragma unroll
    for (int nd = 0; nd < 4; ++nd) {
      v16h vf = cat8(*(const v8h*)&Vt[buf][nd * 16 + lrow][lhalf * 8],
                     *(const v8h*)&Vt[buf][nd * 16 + lrow][lhalf * 8 + 16]);
      o[nd] = wmma16(pf, vf, o[nd]);
    }
    buf ^= 1;
  }

  // normalize and write attention output (f16, feeds out-proj GEMM)
#pragma unroll
  for (int nd = 0; nd < 4; ++nd)
#pragma unroll
    for (int e = 0; e < 8; ++e) {
      const float v = o[nd][e] / lsum[e];
      const size_t row = (size_t)b * SEQ + q0 + 8 * lhalf + e;
      oh[row * EMBED_DIM + h * HEAD_DIM + nd * 16 + lrow] = (_Float16)v;
    }
}

// ---------------------------------------------------------------------------
extern "C" void kernel_launch(void* const* d_in, const int* in_sizes, int n_in,
                              void* d_out, int out_size, void* d_ws, size_t ws_size,
                              hipStream_t stream) {
  const float* x    = (const float*)d_in[0];
  const float* y    = (const float*)d_in[1];
  const float* w_kv = (const float*)d_in[2];
  const float* b_kv = (const float*)d_in[3];
  const float* w_q  = (const float*)d_in[4];
  const float* b_q  = (const float*)d_in[5];
  const float* w_o  = (const float*)d_in[6];
  const float* b_o  = (const float*)d_in[7];
  float* out = (float*)d_out;

  uint8_t* ws = (uint8_t*)d_ws;
  size_t off = 0;
  auto take = [&](size_t elems) -> _Float16* {
    _Float16* p = (_Float16*)(ws + off);
    off += (elems * sizeof(_Float16) + 255) & ~(size_t)255;
    return p;
  };
  _Float16* yh    = take((size_t)BS * CROSS_DIM);            // y in f16
  _Float16* xh    = take((size_t)BS * EMBED_DIM);            // x in f16
  _Float16* wkvh  = take((size_t)2 * EMBED_DIM * CROSS_DIM); // w_kv f16
  _Float16* wqh   = take((size_t)EMBED_DIM * EMBED_DIM);     // w_q f16
  _Float16* woh   = take((size_t)EMBED_DIM * EMBED_DIM);     // w_o f16
  _Float16* kvh   = take((size_t)BS * 2 * EMBED_DIM);        // kv projection
  _Float16* qh    = take((size_t)BS * EMBED_DIM);            // q projection
  _Float16* attnh = take((size_t)BS * EMBED_DIM);            // attention out
  // total ~53 MB of workspace

  auto cvt = [&](const float* src, _Float16* dst, size_t n) {
    int n4 = (int)(n / 4);
    int blocks = (n4 + 255) / 256;
    if (blocks > 4096) blocks = 4096;
    cvt_f32_f16_kernel<<<blocks, 256, 0, stream>>>((const float4*)src, dst, n4);
  };
  cvt(y, yh, (size_t)BS * CROSS_DIM);
  cvt(x, xh, (size_t)BS * EMBED_DIM);
  cvt(w_kv, wkvh, (size_t)2 * EMBED_DIM * CROSS_DIM);
  cvt(w_q, wqh, (size_t)EMBED_DIM * EMBED_DIM);
  cvt(w_o, woh, (size_t)EMBED_DIM * EMBED_DIM);

  // kv = y @ w_kv^T + b_kv   [4096 x 2048], K=768
  gemm_nt_wmma_kernel<true><<<dim3(BS / 128, (2 * EMBED_DIM) / 128), 256, 0,
                              stream>>>(yh, wkvh, b_kv, kvh, nullptr, BS,
                                        2 * EMBED_DIM, CROSS_DIM);
  // q = x @ w_q^T + b_q      [4096 x 1024], K=1024
  gemm_nt_wmma_kernel<true><<<dim3(BS / 128, EMBED_DIM / 128), 256, 0,
                              stream>>>(xh, wqh, b_q, qh, nullptr, BS,
                                        EMBED_DIM, EMBED_DIM);
  // fused multi-head attention -> attnh
  attn_fa_wmma_kernel<<<dim3(SEQ / 128, BATCH * NUM_HEADS), 256, 0, stream>>>(
      qh, kvh, attnh);
  // out = attn @ w_o^T + b_o (f32 final output)
  gemm_nt_wmma_kernel<false><<<dim3(BS / 128, EMBED_DIM / 128), 256, 0,
                               stream>>>(attnh, woh, b_o, nullptr, out, BS,
                                         EMBED_DIM, EMBED_DIM);
}
